// CausalSelfAttention_85134841742028
// MI455X (gfx1250) — compile-verified, matrix operand order checked by reading, not run on hardware
//
#include <hip/hip_runtime.h>

// ---------------- problem constants ----------------
#define Bq   2
#define Tq   2048
#define Cq   2048
#define NHq  16
#define HSq  128
#define NELq 64
#define ATq  10

typedef __bf16 bf16x16 __attribute__((ext_vector_type(16)));
typedef float  f32x8   __attribute__((ext_vector_type(8)));

union FragU {
    bf16x16 v;
    uint4   q[2];
    __bf16  e[16];
};

__device__ __forceinline__ f32x8 zero8() {
    f32x8 z;
#pragma unroll
    for (int i = 0; i < 8; ++i) z[i] = 0.0f;
    return z;
}

// Load a 16x32 bf16 fragment from a row-major tile (ld = row stride in elements).
// Produces the CDNA5 wave32 A/B operand layout:
//   lane = (half<<4)|l16 ; row/col = l16 ; element j <-> K = (j&7) + (j>>3)*16 + half*8
__device__ __forceinline__ bf16x16 load_frag_rm(const __bf16* tile, int ld, int lane) {
    const int l16  = lane & 15;
    const int half = lane >> 4;
    FragU f;
    const __bf16* p = tile + (size_t)l16 * ld + half * 8;
    f.q[0] = *(const uint4*)(p);
    f.q[1] = *(const uint4*)(p + 16);
    return f.v;
}

__device__ __forceinline__ f32x8 wmma_bf16(bf16x16 a, bf16x16 b, f32x8 c) {
    return __builtin_amdgcn_wmma_f32_16x16x32_bf16(false, a, false, b, (short)0, c,
                                                   false, false);
}

// ---------------- elementwise: fp32 -> bf16 ----------------
__global__ void f32_to_bf16(const float* __restrict__ in, __bf16* __restrict__ out, int n) {
    int i = blockIdx.x * blockDim.x + threadIdx.x;
    if (i < n) out[i] = (__bf16)in[i];
}

// ---------------- WMMA NT GEMM: C[m,n] = sum_k A[m,k]*Bw[n,k] + bias[n] ----------------
template <bool BF16OUT>
__global__ void __launch_bounds__(128, 1)
gemm_nt_wmma(const __bf16* __restrict__ A, const __bf16* __restrict__ Bw,
             const float* __restrict__ bias, void* __restrict__ Cout,
             int M, int N, int K) {
    const int lane = threadIdx.x & 31;
    const int wave = blockIdx.x * (blockDim.x >> 5) + (threadIdx.x >> 5);
    const int wn = N >> 6;                 // waves along N (64 cols each)
    const int wm = M >> 5;                 // waves along M (32 rows each)
    if (wave >= wm * wn) return;           // wave-uniform guard (EXEC stays all-1 for WMMA)
    const int m0 = (wave / wn) * 32;
    const int n0 = (wave % wn) * 64;

    f32x8 acc[2][4];
#pragma unroll
    for (int i = 0; i < 2; ++i)
#pragma unroll
        for (int j = 0; j < 4; ++j) acc[i][j] = zero8();

    for (int k0 = 0; k0 < K; k0 += 32) {
        if (k0 + 32 < K) {  // prefetch next K-slab of A rows into cache hierarchy
            __builtin_prefetch(A + (size_t)(m0 + (lane & 31)) * K + k0 + 32, 0, 1);
        }
        bf16x16 a0 = load_frag_rm(A + (size_t)m0 * K + k0, K, lane);
        bf16x16 a1 = load_frag_rm(A + (size_t)(m0 + 16) * K + k0, K, lane);
        bf16x16 bf[4];
#pragma unroll
        for (int j = 0; j < 4; ++j)
            bf[j] = load_frag_rm(Bw + (size_t)(n0 + 16 * j) * K + k0, K, lane);
#pragma unroll
        for (int j = 0; j < 4; ++j) {
            acc[0][j] = wmma_bf16(a0, bf[j], acc[0][j]);
            acc[1][j] = wmma_bf16(a1, bf[j], acc[1][j]);
        }
    }

    const int l16 = lane & 15, half = lane >> 4;
#pragma unroll
    for (int j = 0; j < 4; ++j) {
        const int n = n0 + 16 * j + l16;
        const float bv = bias[n];
#pragma unroll
        for (int i = 0; i < 2; ++i) {
#pragma unroll
            for (int r = 0; r < 8; ++r) {
                const int m = m0 + 16 * i + r + half * 8;
                const float v = acc[i][j][r] + bv;
                if (BF16OUT)
                    ((__bf16*)Cout)[(size_t)m * N + n] = (__bf16)v;
                else
                    ((float*)Cout)[(size_t)m * N + n] = v;
            }
        }
    }
}

// ---------------- RoPE + head split: qkv[B*T,6144] -> Q,K [B,NH,T,128], V^T [B,NH,128,T] ----
__global__ void rope_split(const __bf16* __restrict__ qkvb, const float* __restrict__ cosT,
                           const float* __restrict__ sinT, __bf16* __restrict__ Q,
                           __bf16* __restrict__ Kk, __bf16* __restrict__ VT) {
    const int tid = blockIdx.x * blockDim.x + threadIdx.x;
    if (tid >= Bq * NHq * Tq * HSq) return;
    const int d  = tid & (HSq - 1);
    const int t  = (tid >> 7) & (Tq - 1);
    const int bh = tid >> 18;              // HS*T = 2^18
    const int h  = bh & (NHq - 1);
    const int b  = bh >> 4;

    const size_t src = ((size_t)(b * Tq + t)) * (3 * Cq) + (size_t)h * (3 * HSq);
    float qv = (float)qkvb[src + d];
    float kv = (float)qkvb[src + HSq + d];
    float vv = (float)qkvb[src + 2 * HSq + d];

    if (d < NELq) {
        const float c = cosT[t * NELq + d];
        const float s = sinT[t * NELq + d];
        float qr, kr;
        if (d < NELq / 2) {
            qr = -(float)qkvb[src + d + NELq / 2];
            kr = -(float)qkvb[src + HSq + d + NELq / 2];
        } else {
            qr = (float)qkvb[src + d - NELq / 2];
            kr = (float)qkvb[src + HSq + d - NELq / 2];
        }
        qv = qv * c + qr * s;
        kv = kv * c + kr * s;
    }

    const size_t dst = ((size_t)bh * Tq + t) * HSq + d;
    Q[dst]  = (__bf16)qv;
    Kk[dst] = (__bf16)kv;
    VT[((size_t)bh * HSq + d) * Tq + t] = (__bf16)vv;
}

// ---------------- adapter K/V: ak[NH,16,128] (keys>=10 zero), avT[NH,128,32] (keys>=10 zero) --
__global__ void adapter_kv(const float* __restrict__ wte, const float* __restrict__ attn_w,
                           const float* __restrict__ attn_b, __bf16* __restrict__ AK,
                           __bf16* __restrict__ AVT) {
    const int tid = blockIdx.x * blockDim.x + threadIdx.x;   // NH*128*32 = 65536
    if (tid >= NHq * HSq * 32) return;
    const int j = tid & 31;
    const int d = (tid >> 5) & (HSq - 1);
    const int h = tid >> 12;

    float kvv = 0.0f, vvv = 0.0f;
    if (j < ATq) {
        const float* wr = wte + (size_t)j * Cq;
        const float* wk = attn_w + (size_t)(h * 3 * HSq + HSq + d) * Cq;
        const float* wv = attn_w + (size_t)(h * 3 * HSq + 2 * HSq + d) * Cq;
        float sk = 0.0f, sv = 0.0f;
        for (int c = 0; c < Cq; ++c) {
            const float xv = wr[c];
            sk += xv * wk[c];
            sv += xv * wv[c];
        }
        kvv = sk + attn_b[h * 3 * HSq + HSq + d];
        vvv = sv + attn_b[h * 3 * HSq + 2 * HSq + d];
    }
    if (j < 16) AK[((size_t)h * 16 + j) * HSq + d] = (__bf16)kvv;
    AVT[((size_t)h * HSq + d) * 32 + j] = (__bf16)vvv;
}

// ---------------- flash attention + adapter attention + gating -----------------
// 1 wave per 16-query tile. S^T = K*Q^T so the softmaxed P tile is already in
// B-operand layout for Y^T = V^T * P^T (no LDS transpose of P needed).
// K chunks (32 keys x 128 dims = 8KB) are staged into per-wave LDS double buffers
// with GLOBAL_LOAD_ASYNC_TO_LDS_B128 (ASYNCcnt) so chunk i+1 streams in while the
// 8 S-WMMAs consume chunk i from LDS (ds_load_b128, DScnt).
__global__ void __launch_bounds__(128, 1)
attn_fused(const __bf16* __restrict__ Q, const __bf16* __restrict__ Kk,
           const __bf16* __restrict__ VT, const __bf16* __restrict__ AK,
           const __bf16* __restrict__ AVT, const float* __restrict__ gating,
           __bf16* __restrict__ Yout) {
    __shared__ __bf16 ldsK[4][2][32 * HSq];   // 4 waves x 2 buffers x 8KB = 64KB
    const int lane = threadIdx.x & 31;
    const int wv   = threadIdx.x >> 5;        // wave within workgroup
    const int wave = blockIdx.x * (blockDim.x >> 5) + wv;
    const int QT = Tq / 16;
    const int qt = wave % QT;
    const int bh = wave / QT;              // [0, B*NH)
    const int h  = bh & (NHq - 1);
    const int b  = bh >> 4;
    const int q0 = qt * 16;

    const __bf16* qbase  = Q  + (size_t)bh * Tq * HSq;
    const __bf16* kbase  = Kk + (size_t)bh * Tq * HSq;
    const __bf16* vtbase = VT + (size_t)bh * HSq * Tq;

    const int l16 = lane & 15, half = lane >> 4;
    const int qg = q0 + l16;               // this lane's query index (its D-tile column)
    const float scale = 0.08838834764831845f;   // 1/sqrt(128)

    // async-stage one 32x128 bf16 K chunk into this wave's LDS buffer (16 x 512B)
    auto stage_k = [&](int buf, int kk) {
        const char*    g = (const char*)(kbase + (size_t)kk * HSq);
        const uint32_t l = (uint32_t)(uintptr_t)&ldsK[wv][buf][0];
#pragma unroll
        for (int i = 0; i < 16; ++i) {
            uint64_t ga = (uint64_t)g + (uint32_t)(i * 512 + lane * 16);
            uint32_t la = l + (uint32_t)(i * 512 + lane * 16);
            asm volatile("global_load_async_to_lds_b128 %0, %1, off"
                         :: "v"(la), "v"(ga) : "memory");
        }
    };

    // Q as B-operand fragments (4 x 32-dim windows), loaded once
    bf16x16 qf[4];
#pragma unroll
    for (int dd = 0; dd < 4; ++dd)
        qf[dd] = load_frag_rm(qbase + (size_t)q0 * HSq + 32 * dd, HSq, lane);

    f32x8 yacc[8];
#pragma unroll
    for (int g = 0; g < 8; ++g) yacc[g] = zero8();
    float m_run = -1e30f, l_run = 0.0f;

    const int nchunks = (q0 + 16 + 31) / 32;
    stage_k(0, 0);
    for (int ci = 0; ci < nchunks; ++ci) {
        const int kk  = ci * 32;
        const int cur = ci & 1;
        if (ci + 1 < nchunks) {
            // prior DS reads of the other buffer must be done before overwriting it
            asm volatile("s_wait_dscnt 0x0" ::: "memory");
            stage_k(cur ^ 1, kk + 32);
            // cur buffer complete once ASYNCcnt <= 16 (next chunk's 16 still in flight)
            asm volatile("s_wait_asynccnt 0x10" ::: "memory");
        } else {
            asm volatile("s_wait_asynccnt 0x0" ::: "memory");
        }
        const __bf16* kc = &ldsK[wv][cur][0];
        bf16x16 kf[8];
#pragma unroll
        for (int dd = 0; dd < 4; ++dd) {
            kf[dd]     = load_frag_rm(kc + 32 * dd, HSq, lane);
            kf[4 + dd] = load_frag_rm(kc + 16 * HSq + 32 * dd, HSq, lane);
        }
        f32x8 slo = zero8(), shi = zero8();
#pragma unroll
        for (int dd = 0; dd < 4; ++dd) {
            slo = wmma_bf16(kf[dd], qf[dd], slo);
            shi = wmma_bf16(kf[4 + dd], qf[dd], shi);
        }
        // scale + causal mask; S^T tile rows = keys (r + half*8 [+16]), col = query l16
        float mx = -1e30f;
#pragma unroll
        for (int r = 0; r < 8; ++r) {
            float s = slo[r] * scale;
            if (kk + r + half * 8 > qg) s = -1e30f;
            slo[r] = s;
            mx = fmaxf(mx, s);
            s = shi[r] * scale;
            if (kk + 16 + r + half * 8 > qg) s = -1e30f;
            shi[r] = s;
            mx = fmaxf(mx, s);
        }
        mx = fmaxf(mx, __shfl_xor(mx, 16, 32));      // other half-lane holds other 16 keys
        const float mnew  = fmaxf(m_run, mx);
        const float alpha = __expf(m_run - mnew);
        float lsum = 0.0f;
        FragU pf;                                    // P^T directly in B-operand layout
#pragma unroll
        for (int r = 0; r < 8; ++r) {
            const float p0 = __expf(slo[r] - mnew);
            lsum += p0;
            pf.e[r] = (__bf16)p0;
            const float p1 = __expf(shi[r] - mnew);
            lsum += p1;
            pf.e[8 + r] = (__bf16)p1;
        }
        lsum += __shfl_xor(lsum, 16, 32);
        l_run = l_run * alpha + lsum;
        m_run = mnew;
#pragma unroll
        for (int g = 0; g < 8; ++g) yacc[g] = yacc[g] * alpha;
        // ---- P*V from global (V^T rows are contiguous); batches of 4 ----
#pragma unroll
        for (int gb = 0; gb < 2; ++gb) {
            bf16x16 vf[4];
#pragma unroll
            for (int g = 0; g < 4; ++g)
                vf[g] = load_frag_rm(vtbase + (size_t)((gb * 4 + g) * 16) * Tq + kk, Tq, lane);
#pragma unroll
            for (int g = 0; g < 4; ++g)
                yacc[gb * 4 + g] = wmma_bf16(vf[g], pf.v, yacc[gb * 4 + g]);
        }
    }

    // -------- adapter attention (10 keys, no mask, own softmax) --------
    const __bf16* akb  = AK  + (size_t)h * 16 * HSq;
    const __bf16* avtb = AVT + (size_t)h * HSq * 32;
    f32x8 sa = zero8();
    {
        bf16x16 af[4];
#pragma unroll
        for (int dd = 0; dd < 4; ++dd)
            af[dd] = load_frag_rm(akb + 32 * dd, HSq, lane);
#pragma unroll
        for (int dd = 0; dd < 4; ++dd)
            sa = wmma_bf16(af[dd], qf[dd], sa);
    }

    float mxa = -1e30f;
    float sav[8];
#pragma unroll
    for (int r = 0; r < 8; ++r) {
        float s = sa[r] * scale;
        if (r + half * 8 >= ATq) s = -1e30f;         // pad keys 10..15
        sav[r] = s;
        mxa = fmaxf(mxa, s);
    }
    mxa = fmaxf(mxa, __shfl_xor(mxa, 16, 32));
    float suma = 0.0f;
    FragU paf;
#pragma unroll
    for (int r = 0; r < 8; ++r) {
        const float p = __expf(sav[r] - mxa);
        suma += p;
        paf.e[r]     = (__bf16)p;
        paf.e[8 + r] = (__bf16)0.0f;                 // keys 16..31 absent
    }
    suma += __shfl_xor(suma, 16, 32);

    const float gf    = gating[0];
    const float inv_l = 1.0f / l_run;
    const float inv_a = gf / suma;

    // combine + write: row t = q0+l16, cols h*128 + g*16 + half*8 + r (8 contiguous bf16)
    __bf16* yb = Yout + ((size_t)(b * Tq + q0 + l16)) * Cq + h * HSq;
#pragma unroll
    for (int g = 0; g < 8; ++g) {
        f32x8 ya = wmma_bf16(load_frag_rm(avtb + (size_t)(g * 16) * 32, 32, lane),
                             paf.v, zero8());
        union { __bf16 e[8]; uint4 u; } outv;
#pragma unroll
        for (int r = 0; r < 8; ++r)
            outv.e[r] = (__bf16)(yacc[g][r] * inv_l + ya[r] * inv_a);
        *(uint4*)(yb + g * 16 + half * 8) = outv.u;
    }
}

// ---------------- host launch ----------------
extern "C" void kernel_launch(void* const* d_in, const int* in_sizes, int n_in,
                              void* d_out, int out_size, void* d_ws, size_t ws_size,
                              hipStream_t stream) {
    const float* x      = (const float*)d_in[0];
    const float* cosT   = (const float*)d_in[1];
    const float* sinT   = (const float*)d_in[2];
    // d_in[3] = causal mask (bool) -- causality computed analytically, unused
    const float* attn_w = (const float*)d_in[4];
    const float* attn_b = (const float*)d_in[5];
    const float* proj_w = (const float*)d_in[6];
    const float* proj_b = (const float*)d_in[7];
    const float* wte    = (const float*)d_in[8];
    const float* gating = (const float*)d_in[9];

    char*  ws  = (char*)d_ws;
    size_t off = 0;
    auto alloc = [&](size_t bytes) -> char* {
        char* p = ws + off;
        off += (bytes + 255) & ~(size_t)255;
        return p;
    };

    const size_t nX   = (size_t)Bq * Tq * Cq;        // 8.4M
    const size_t nW   = (size_t)3 * Cq * Cq;         // 12.6M
    const size_t nQKV = (size_t)Bq * Tq * 3 * Cq;    // 25.2M
    const size_t nH   = (size_t)Bq * NHq * Tq * HSq; // 8.4M
    const size_t nPW  = (size_t)Cq * Cq;             // 4.2M

    __bf16* xb    = (__bf16*)alloc(nX * 2);
    __bf16* wb    = (__bf16*)alloc(nW * 2);
    __bf16* qkvb  = (__bf16*)alloc(nQKV * 2);
    __bf16* qbuf  = (__bf16*)alloc(nH * 2);
    __bf16* kbuf  = (__bf16*)alloc(nH * 2);
    __bf16* vtbuf = (__bf16*)alloc(nH * 2);
    __bf16* akb   = (__bf16*)alloc((size_t)NHq * 16 * HSq * 2);
    __bf16* avtb  = (__bf16*)alloc((size_t)NHq * HSq * 32 * 2);
    __bf16* ycomb = (__bf16*)alloc(nX * 2);
    __bf16* pwb   = (__bf16*)alloc(nPW * 2);

    // 1) fp32 -> bf16 conversions
    f32_to_bf16<<<(int)((nX + 255) / 256), 256, 0, stream>>>(x, xb, (int)nX);
    f32_to_bf16<<<(int)((nW + 255) / 256), 256, 0, stream>>>(attn_w, wb, (int)nW);
    f32_to_bf16<<<(int)((nPW + 255) / 256), 256, 0, stream>>>(proj_w, pwb, (int)nPW);

    // 2) QKV GEMM: [4096,2048] x [6144,2048]^T  (12288 waves, 4 waves/WG)
    {
        const int M = Bq * Tq, N = 3 * Cq, K = Cq;
        const int waves = (M / 32) * (N / 64);
        gemm_nt_wmma<true><<<waves / 4, 128, 0, stream>>>(xb, wb, attn_b, qkvb, M, N, K);
    }

    // 3) RoPE + head split (+ V transpose)
    rope_split<<<(int)((nH + 255) / 256), 256, 0, stream>>>(qkvb, cosT, sinT, qbuf, kbuf, vtbuf);

    // 4) adapter K/V (tiny)
    adapter_kv<<<(NHq * HSq * 32) / 256, 256, 0, stream>>>(wte, attn_w, attn_b, akb, avtb);

    // 5) fused flash attention + adapter + gating (4096 waves, 4 waves/WG)
    {
        const int waves = Bq * NHq * (Tq / 16);
        attn_fused<<<waves / 4, 128, 0, stream>>>(qbuf, kbuf, vtbuf, akb, avtb, gating, ycomb);
    }

    // 6) output projection: [4096,2048] x [2048,2048]^T -> fp32 d_out
    {
        const int M = Bq * Tq, N = Cq, K = Cq;
        const int waves = (M / 32) * (N / 64);
        gemm_nt_wmma<false><<<waves / 4, 128, 0, stream>>>(ycomb, pwb, proj_b, (float*)d_out,
                                                           M, N, K);
    }
    (void)in_sizes; (void)n_in; (void)out_size; (void)ws_size;
}
